// ConcatLSTM_57062935494750
// MI455X (gfx1250) — matrix operand order repaired and checked by reading, not compiled, not run on hardware
//
#include <hip/hip_runtime.h>
#include <hip/hip_bf16.h>

// ---------------------------------------------------------------------------
// Problem constants (match reference)
// ---------------------------------------------------------------------------
constexpr int T_ = 512;
constexpr int B_ = 64;
constexpr int I_ = 256;
constexpr int H_ = 512;
constexpr int C_ = 64;
constexpr int G_ = 4 * H_;        // 2048

typedef __bf16 bf16_t;
typedef __attribute__((ext_vector_type(16))) __bf16 v16bf;
typedef __attribute__((ext_vector_type(8)))  float  v8f;
typedef __attribute__((ext_vector_type(4)))  unsigned int u32x4;
typedef __attribute__((ext_vector_type(8)))  int i32x8;
typedef __attribute__((ext_vector_type(4)))  int i32x4;

union FragU { v16bf v; uint4 q[2]; };

// ---------------------------------------------------------------------------
// WMMA fragment loaders (layouts per CDNA5 ISA, wave32)
// ---------------------------------------------------------------------------
// A-matrix 16x32 bf16: lane<16 -> row=lane, K chunks {0..7} (v0..3), {16..23} (v4..7)
//                      lane>=16 -> row=lane-16, K chunks {8..15}, {24..31}
__device__ __forceinline__ v16bf load_frag_a(const bf16_t* __restrict__ tile,
                                             int ld, int lane) {
    const int row = lane & 15;
    const int kb  = (lane >> 4) ? 8 : 0;
    const bf16_t* p = tile + (size_t)row * ld + kb;
    FragU u;
    u.q[0] = *(const uint4*)(p);        // K = kb .. kb+7
    u.q[1] = *(const uint4*)(p + 16);   // K = kb+16 .. kb+23
    return u.v;
}

// B-matrix 32x16 bf16 stored as rows of W (row = output column n, contiguous K):
// lanes 0-15 hold K={0..7,16..23} of column n, lanes 16-31 hold K={8..15,24..31}.
__device__ __forceinline__ v16bf load_frag_b(const bf16_t* __restrict__ tile,
                                             int ld, int lane) {
    const int n  = lane & 15;
    const int kb = (lane >> 4) ? 8 : 0;
    const bf16_t* p = tile + (size_t)n * ld + kb;
    FragU u;
    u.q[0] = *(const uint4*)(p);
    u.q[1] = *(const uint4*)(p + 16);
    return u.v;
}

// Branch-free activations (one v_exp_f32 each; saturate correctly at +-inf)
__device__ __forceinline__ float sigmoidf_(float x) {
    return 1.0f / (1.0f + __expf(-x));
}
__device__ __forceinline__ float tanhf_(float x) {
    return 1.0f - 2.0f / (__expf(2.0f * x) + 1.0f);
}

// ---------------------------------------------------------------------------
// Grid-wide barrier (persistent kernel, epoch-based)
// ---------------------------------------------------------------------------
__device__ __forceinline__ void grid_sync(unsigned* cnt, unsigned* epoch,
                                          unsigned nblocks, unsigned e) {
    __threadfence();
    __syncthreads();
    if (threadIdx.x == 0) {
        if (atomicAdd(cnt, 1u) == nblocks - 1u) {
            atomicExch(cnt, 0u);
            __hip_atomic_store(epoch, e, __ATOMIC_RELEASE, __HIP_MEMORY_SCOPE_AGENT);
        } else {
            while (__hip_atomic_load(epoch, __ATOMIC_ACQUIRE,
                                     __HIP_MEMORY_SCOPE_AGENT) < e) {
                __builtin_amdgcn_s_sleep(2);
            }
        }
    }
    __syncthreads();
}

// ---------------------------------------------------------------------------
// TDM: stage h_prev [64 x 512] bf16 from global into LDS (D# per ISA ch.8)
// ---------------------------------------------------------------------------
__device__ __forceinline__ void tdm_load_h(const bf16_t* hprev, bf16_t* hlds) {
    const unsigned long long ga = (unsigned long long)(uintptr_t)hprev;
    const unsigned ldsa = (unsigned)(uintptr_t)hlds;     // addr[31:0] = LDS offset

    u32x4 g0;
    g0[0] = 1u;                                          // count=1, user descriptor
    g0[1] = ldsa;                                        // lds_addr   [63:32]
    g0[2] = (unsigned)ga;                                // global_addr[95:64]
    g0[3] = ((unsigned)(ga >> 32) & 0x01FFFFFFu)         // global_addr[120:96]
          | (2u << 30);                                  // type=2 (image) [127:126]

    i32x8 g1;
    g1[0] = 0x00010000;            // workgroup_mask=0, data_size=1 (2 bytes)
    g1[1] = (int)(512u << 16);     // tensor_dim0[15:0] = 512 (K elems)
    g1[2] = (int)(64u  << 16);     // tensor_dim0 hi = 0 ; tensor_dim1 = 64 rows
    g1[3] = (int)(512u << 16);     // tensor_dim1 hi = 0 ; tile_dim0 = 512
    g1[4] = 64;                    // tile_dim1 = 64 ; tile_dim2 = 0
    g1[5] = 512;                   // tensor_dim0_stride lo = 512
    g1[6] = 0;                     // stride hi = 0
    g1[7] = 0;
    const i32x4 z4 = {0, 0, 0, 0};                       // 2-D tensor: groups 2/3 unused
#if defined(__clang_major__) && (__clang_major__ >= 23)
    const i32x8 z8 = {0, 0, 0, 0, 0, 0, 0, 0};
    __builtin_amdgcn_tensor_load_to_lds(g0, g1, z4, z4, z8, 0);
#else
    __builtin_amdgcn_tensor_load_to_lds(g0, g1, z4, z4, 0);
#endif
}

// ---------------------------------------------------------------------------
// Small prep kernels
// ---------------------------------------------------------------------------
__global__ void init_kernel(unsigned* __restrict__ sync, bf16_t* __restrict__ hbuf) {
    if (blockIdx.x == 0 && threadIdx.x < 2) sync[threadIdx.x] = 0u;
    const int stride = gridDim.x * blockDim.x;
    for (int i = blockIdx.x * blockDim.x + threadIdx.x; i < 2 * B_ * H_; i += stride)
        hbuf[i] = (bf16_t)0.0f;
}

__global__ void cvt_bf16_kernel(const float* __restrict__ in,
                                bf16_t* __restrict__ out, size_t n) {
    const size_t stride = (size_t)gridDim.x * blockDim.x;
    for (size_t i = (size_t)blockIdx.x * blockDim.x + threadIdx.x; i < n; i += stride)
        out[i] = (bf16_t)in[i];
}

// base[b,g] = bias[g] + sum_c ctx[b,c] * w_mh[g,c]
__global__ void base_kernel(const float* __restrict__ ctx,
                            const float* __restrict__ w_mh,
                            const float* __restrict__ bias,
                            float* __restrict__ out) {
    const int idx = blockIdx.x * blockDim.x + threadIdx.x;   // over B_*G_
    if (idx >= B_ * G_) return;
    const int b = idx / G_;
    const int g = idx % G_;
    float acc = bias[g];
    #pragma unroll 8
    for (int c = 0; c < C_; ++c)
        acc = fmaf(ctx[b * C_ + c], w_mh[g * C_ + c], acc);
    out[idx] = acc;
}

// ---------------------------------------------------------------------------
// Persistent recurrent LSTM layer.
// 32 blocks x 128 threads (4 waves). Block owns hidden slice j0..j0+15 (its
// four gate columns i/f/g/o are four aligned 16-wide N tiles). Wave w owns
// batch rows 16w..16w+15. gates = [x_t | h_{t-1}] @ [w_ih | w_hh]^T + base.
// Weights stationary in LDS; h_prev staged into LDS by the TDM each step,
// overlapped with the x-projection WMMAs.
// ---------------------------------------------------------------------------
__global__ __launch_bounds__(128)
void lstm_layer_kernel(const bf16_t* __restrict__ xin, int Kx,
                       const float* __restrict__ w_ih,   // [G, Kx]
                       const float* __restrict__ w_hh,   // [G, H]
                       const float* __restrict__ base,   // [B, G]
                       bf16_t* __restrict__ hbuf,        // [2][B][H] ping-pong
                       bf16_t* __restrict__ ybf,         // [T][B][H] or null
                       float*  __restrict__ yf32,        // [T][B][H] or null
                       unsigned* __restrict__ sync) {
    extern __shared__ bf16_t smem[];
    const int KT   = Kx + H_;
    bf16_t* wlds = smem;                                  // [64][KT] row-major
    bf16_t* hlds = smem + (size_t)64 * KT;                // [64][H]  h_prev stage
    const int j0   = blockIdx.x * 16;
    const int tid  = threadIdx.x;
    const int lane = tid & 31;
    const int wave = tid >> 5;
    const int m0   = wave * 16;
    unsigned* cnt   = sync;
    unsigned* epoch = sync + 1;

    // ---- Stage concatenated weight slice [w_ih | w_hh] -> LDS (f32 -> bf16)
    for (int idx = tid; idx < 64 * KT; idx += blockDim.x) {
        const int rl = idx / KT;
        const int k  = idx - rl * KT;
        const int g  = (rl >> 4) * H_ + j0 + (rl & 15);
        const float wv = (k < Kx) ? w_ih[(size_t)g * Kx + k]
                                  : w_hh[(size_t)g * H_ + (k - Kx)];
        wlds[idx] = (bf16_t)wv;
    }
    __syncthreads();

    // ---- Preload base (bias + ctx proj) tiles in accumulator layout
    const int n  = lane & 15;
    const int mh = (lane >> 4) * 8;
    v8f basef[4];
    #pragma unroll
    for (int q = 0; q < 4; ++q)
        #pragma unroll
        for (int r = 0; r < 8; ++r)
            basef[q][r] = base[(size_t)(m0 + mh + r) * G_ + q * H_ + j0 + n];

    v8f cst = {0.f, 0.f, 0.f, 0.f, 0.f, 0.f, 0.f, 0.f};   // cell-state tile

    const int KXSTEPS = Kx / 32;
    const int KHSTEPS = H_ / 32;

    for (int t = 0; t < T_; ++t) {
        const bf16_t* hprev = hbuf + (size_t)(t & 1) * B_ * H_;
        bf16_t*       hnext = hbuf + (size_t)((t + 1) & 1) * B_ * H_;

        // Kick the TDM: h_prev -> LDS, overlapped with the x-phase below.
        if (wave == 0) tdm_load_h(hprev, hlds);

        v8f acc[4] = {basef[0], basef[1], basef[2], basef[3]};

        // ---- x phase: A fragments from global bf16 x_t
        const bf16_t* xt = xin + ((size_t)t * B_ + m0) * Kx;
        for (int kk = 0; kk < KXSTEPS; ++kk) {
            const v16bf a = load_frag_a(xt + kk * 32, Kx, lane);
            #pragma unroll
            for (int q = 0; q < 4; ++q) {
                const v16bf b = load_frag_b(wlds + (size_t)(q * 16) * KT + kk * 32,
                                            KT, lane);
                acc[q] = __builtin_amdgcn_wmma_f32_16x16x32_bf16(
                    false, a, false, b, (short)0, acc[q], false, false);
            }
        }

        // Prefetch next step's x tile (global_prefetch_b8)
        if (t + 1 < T_) {
            const bf16_t* xn = xin + ((size_t)(t + 1) * B_ + m0 + (lane & 15)) * Kx
                             + (lane >> 4) * (Kx >> 1);
            __builtin_prefetch((const void*)xn, 0, 1);
        }

        // ---- wait for the TDM, publish h_prev tile block-wide
        if (wave == 0) __builtin_amdgcn_s_wait_tensorcnt(0);
        __syncthreads();

        // ---- h phase: A fragments from LDS-staged h_prev
        const bf16_t* hA = hlds + (size_t)m0 * H_;
        for (int kh = 0; kh < KHSTEPS; ++kh) {
            const v16bf a = load_frag_a(hA + kh * 32, H_, lane);
            #pragma unroll
            for (int q = 0; q < 4; ++q) {
                const v16bf b = load_frag_b(wlds + (size_t)(q * 16) * KT + Kx + kh * 32,
                                            KT, lane);
                acc[q] = __builtin_amdgcn_wmma_f32_16x16x32_bf16(
                    false, a, false, b, (short)0, acc[q], false, false);
            }
        }

        // ---- LSTM elementwise in accumulator layout (gates share lane map)
        float hv[8];
        #pragma unroll
        for (int r = 0; r < 8; ++r) {
            const float ig = sigmoidf_(acc[0][r]);
            const float fg = sigmoidf_(acc[1][r]);
            const float gg = tanhf_(acc[2][r]);
            const float og = sigmoidf_(acc[3][r]);
            const float c  = fg * cst[r] + ig * gg;
            cst[r] = c;
            hv[r]  = og * tanhf_(c);
        }

        // ---- Scatter h tile: (m, n) -> vgpr r = m&7, lane = n + 16*(m>>3)
        #pragma unroll
        for (int r = 0; r < 8; ++r) {
            const int m = m0 + mh + r;
            const size_t ho = (size_t)m * H_ + j0 + n;
            const bf16_t hb = (bf16_t)hv[r];
            hnext[ho] = hb;
            if (ybf)  ybf [(size_t)t * B_ * H_ + ho] = hb;
            if (yf32) yf32[(size_t)t * B_ * H_ + ho] = hv[r];
        }

        grid_sync(cnt, epoch, gridDim.x, (unsigned)(t + 1));
    }
}

// ---------------------------------------------------------------------------
// Host launcher
// ---------------------------------------------------------------------------
extern "C" void kernel_launch(void* const* d_in, const int* in_sizes, int n_in,
                              void* d_out, int out_size, void* d_ws, size_t ws_size,
                              hipStream_t stream) {
    const float* x     = (const float*)d_in[0];
    const float* ctx   = (const float*)d_in[1];
    const float* w_ih0 = (const float*)d_in[2];
    const float* w_hh0 = (const float*)d_in[3];
    const float* w_mh0 = (const float*)d_in[4];
    const float* b0    = (const float*)d_in[5];
    const float* w_ih1 = (const float*)d_in[6];
    const float* w_hh1 = (const float*)d_in[7];
    const float* w_mh1 = (const float*)d_in[8];
    const float* b1    = (const float*)d_in[9];
    float* out = (float*)d_out;

    char* ws = (char*)d_ws;
    size_t off = 0;
    auto wsalloc = [&](size_t bytes) -> void* {
        void* p = ws + off;
        off = (off + bytes + 255) & ~(size_t)255;
        return p;
    };
    unsigned* sync  = (unsigned*)wsalloc(2 * sizeof(unsigned));
    bf16_t*   hbuf  = (bf16_t*)  wsalloc((size_t)2 * B_ * H_ * sizeof(bf16_t));
    bf16_t*   xbf   = (bf16_t*)  wsalloc((size_t)T_ * B_ * I_ * sizeof(bf16_t));
    bf16_t*   y0    = (bf16_t*)  wsalloc((size_t)T_ * B_ * H_ * sizeof(bf16_t));
    float*    base0 = (float*)   wsalloc((size_t)B_ * G_ * sizeof(float));
    float*    base1 = (float*)   wsalloc((size_t)B_ * G_ * sizeof(float));

    // 1) convert x to bf16
    const size_t nx = (size_t)T_ * B_ * I_;
    cvt_bf16_kernel<<<2048, 256, 0, stream>>>(x, xbf, nx);

    // 2) ctx projections + bias -> base tensors
    base_kernel<<<(B_ * G_ + 255) / 256, 256, 0, stream>>>(ctx, w_mh0, b0, base0);
    base_kernel<<<(B_ * G_ + 255) / 256, 256, 0, stream>>>(ctx, w_mh1, b1, base1);

    // 3) layer 0: Kx = I (256); output bf16 sequence y0
    init_kernel<<<64, 256, 0, stream>>>(sync, hbuf);
    {
        const size_t shmem = ((size_t)64 * (I_ + H_) + (size_t)64 * H_)
                           * sizeof(bf16_t);                       // 160 KB
        lstm_layer_kernel<<<32, 128, shmem, stream>>>(
            xbf, I_, w_ih0, w_hh0, base0, hbuf, y0, nullptr, sync);
    }

    // 4) layer 1: Kx = H (512); output f32 -> d_out
    init_kernel<<<64, 256, 0, stream>>>(sync, hbuf);
    {
        const size_t shmem = ((size_t)64 * (H_ + H_) + (size_t)64 * H_)
                           * sizeof(bf16_t);                       // 192 KB
        lstm_layer_kernel<<<32, 128, shmem, stream>>>(
            y0, H_, w_ih1, w_hh1, base1, hbuf, nullptr, out, sync);
    }
}
